// Cross_TransformerBlock_v2_30923764531524
// MI455X (gfx1250) — matrix operand order repaired and checked by reading, not compiled, not run on hardware
//
#include <hip/hip_runtime.h>
#include <hip/hip_bf16.h>

// ---------------------------------------------------------------------------
// Cross_TransformerBlock_v2 for MI455X (gfx1250, wave32, WMMA + TDM)
//   B=2, H=W=256, C=64, NH=4, G=8  -> 2048 windows of N=128 tokens, d=16
//   1) posbias_kernel : 465x4 dynamic position-bias MLP table -> d_ws
//   2) attn_kernel    : per-window fused LN1 + QKV + (QK^T + bias) softmax
//                       + PV + proj + residual  (f16 WMMA, f32 accum)
//                       activation tile pre-staged to LDS by TENSOR_LOAD_TO_LDS
//   3) ffn_kernel     : per-128-token fused LN2 + 64->256 GELU -> 64 + residual
// ---------------------------------------------------------------------------

typedef __attribute__((ext_vector_type(16))) _Float16 v16h;
typedef __attribute__((ext_vector_type(8)))  _Float16 v8h;
typedef __attribute__((ext_vector_type(8)))  float    v8f;
typedef __attribute__((ext_vector_type(4)))  unsigned int u32x4;
typedef __attribute__((ext_vector_type(8)))  int          i32x8;
typedef __attribute__((ext_vector_type(4)))  int          i32x4;

#define NWIN   2048              // B * 32 * 32 windows
#define OUTHALF 8388608ULL       // B*H*W*C elements per modality

#if defined(__has_builtin)
#if __has_builtin(__builtin_amdgcn_tensor_load_to_lds)
#define HAVE_TDM 1
#endif
#endif
#ifndef HAVE_TDM
#define HAVE_TDM 0
#endif

// ---- WMMA helpers ---------------------------------------------------------

__device__ __forceinline__ v8f wmma_f16(v16h a, v16h b, v8f c) {
  return __builtin_amdgcn_wmma_f32_16x16x32_f16(false, a, false, b,
                                                (short)0, c, false, false);
}

// A fragment: 16x32 f16 tile at (m0,k0) of row-major matrix Mp (ld halves).
// ISA 7.12.2: row = m0 + (lane&15), ab = (lane>=16)?8:0,
//   vgpr j<4 -> K pair (ab+2j), j>=4 -> K pair (16+ab+2(j-4)). kw zero-pads.
__device__ __forceinline__ v16h frag_a(const _Float16* Mp, int ld, int m0,
                                       int k0, int kw, int lane) {
  int row = m0 + (lane & 15);
  int ab  = (lane & 16) ? 8 : 0;
  const _Float16* p = Mp + row * ld + k0;
  v16h f;
#pragma unroll
  for (int j = 0; j < 8; ++j) {
    int k = (j < 4) ? (ab + 2 * j) : (16 + ab + 2 * (j - 4));
    float w = (k < kw) ? *(const float*)(p + k) : 0.f;
    union { float f32; _Float16 h[2]; } u; u.f32 = w;
    f[2 * j]     = u.h[0];
    f[2 * j + 1] = u.h[1];
  }
  return f;
}

// B fragment: 32x16 tile of B (KxN) supplied as Bt (NxK row-major, ld halves),
// tile origin (k0,n0). Lane: n = n0+(lane&15), kb=(lane>=16)?8:0,
//   j<4 -> pair kb+2j, j>=4 -> pair 16+kb+2(j-4).
__device__ __forceinline__ v16h frag_b(const _Float16* Bt, int ld, int n0,
                                       int k0, int kw, int lane) {
  int row = n0 + (lane & 15);
  int kb  = (lane & 16) ? 8 : 0;
  const _Float16* p = Bt + row * ld + k0;
  v16h f;
#pragma unroll
  for (int j = 0; j < 8; ++j) {
    int k = (j < 4) ? (kb + 2 * j) : (16 + kb + 2 * (j - 4));
    float w = (k < kw) ? *(const float*)(p + k) : 0.f;
    union { float f32; _Float16 h[2]; } u; u.f32 = w;
    f[2 * j]     = u.h[0];
    f[2 * j + 1] = u.h[1];
  }
  return f;
}

// ---- TDM: async 2D f32 tile load, global -> LDS ---------------------------
// Builds a D# per CDNA5 ISA and issues
// TENSOR_LOAD_TO_LDS. 2D tile: tile1 rows of tile0 elems, row stride `stride0`
// (all counts in f32 elements). lds_addr = LDS byte offset of destination.
#if HAVE_TDM
__device__ __forceinline__ void tdm_load_2d(unsigned lds_addr,
                                            unsigned long long gaddr,
                                            unsigned tile0, unsigned tile1,
                                            unsigned long long stride0) {
  const unsigned td0 = 1u << 24, td1 = 1u << 24;  // tensor dims (OOB-safe)
  u32x4 g0;
  g0[0] = 1u;                                        // count=1, user D#
  g0[1] = lds_addr;                                  // lds_addr[31:0]
  g0[2] = (unsigned)(gaddr & 0xFFFFFFFFull);         // global_addr[31:0]
  g0[3] = (unsigned)((gaddr >> 32) & 0x1FFFFFFull)   // global_addr[56:32]
          | (2u << 30);                              // type=2 ("image")
  i32x8 g1;
  g1[0] = (int)(2u << 16);                           // data_size=4B
  g1[1] = (int)((td0 & 0xFFFFu) << 16);              // tensor_dim0[15:0]
  g1[2] = (int)((td0 >> 16) | ((td1 & 0xFFFFu) << 16));
  g1[3] = (int)((td1 >> 16) | (tile0 << 16));        // tile_dim0
  g1[4] = (int)(tile1 & 0xFFFFu);                    // tile_dim1 (tile_dim2=0)
  g1[5] = (int)(stride0 & 0xFFFFFFFFull);            // dim0_stride[31:0]
  g1[6] = (int)((stride0 >> 32) & 0xFFFFull);        // dim0_stride[47:32]
  g1[7] = 0;
  i32x4 z4 = {0, 0, 0, 0};
#if __clang_major__ >= 23
  i32x8 z8 = {0, 0, 0, 0, 0, 0, 0, 0};
  __builtin_amdgcn_tensor_load_to_lds(g0, g1, z4, z4, z8, 0);
#else
  __builtin_amdgcn_tensor_load_to_lds(g0, g1, z4, z4, 0);
#endif
}
__device__ __forceinline__ unsigned lds_offset_of(const void* p) {
  return (unsigned)((unsigned long long)(uintptr_t)p & 0x7FFFFull);  // <512KB
}
#endif

// ---- Kernel 1: dynamic position-bias MLP table ----------------------------

__device__ __forceinline__ void ln_relu_mm4(const float* p, const float* g,
                                            const float* b, const float* w,
                                            const float* wb, float* out) {
  float mu = 0.25f * (p[0] + p[1] + p[2] + p[3]);
  float var = 0.f;
#pragma unroll
  for (int i = 0; i < 4; ++i) { float d = p[i] - mu; var += d * d; }
  var *= 0.25f;
  float inv = rsqrtf(var + 1e-5f);
  float r[4];
#pragma unroll
  for (int i = 0; i < 4; ++i) {
    float v = (p[i] - mu) * inv * g[i] + b[i];
    r[i] = v > 0.f ? v : 0.f;
  }
#pragma unroll
  for (int j = 0; j < 4; ++j) {
    float s = wb[j];
#pragma unroll
    for (int i = 0; i < 4; ++i) s += r[i] * w[i * 4 + j];
    out[j] = s;
  }
}

__global__ __launch_bounds__(512) void posbias_kernel(
    const float* pp_w, const float* pp_b,
    const float* ln1g, const float* ln1b, const float* l1w, const float* l1b,
    const float* ln2g, const float* ln2b, const float* l2w, const float* l2b,
    const float* ln3g, const float* ln3b, const float* l3w, const float* l3b,
    float* ptab) {
  int t = blockIdx.x * blockDim.x + threadIdx.x;
  if (t >= 465) return;
  float bx = (float)(t / 15 - 15);  // dh in [-15, 15]
  float by = (float)(t % 15 - 7);   // dw in [-7, 7]
  float p[4], q[4];
#pragma unroll
  for (int j = 0; j < 4; ++j) p[j] = bx * pp_w[j] + by * pp_w[4 + j] + pp_b[j];
  ln_relu_mm4(p, ln1g, ln1b, l1w, l1b, q);
  ln_relu_mm4(q, ln2g, ln2b, l2w, l2b, p);
  ln_relu_mm4(p, ln3g, ln3b, l3w, l3b, q);
#pragma unroll
  for (int h = 0; h < 4; ++h) ptab[t * 4 + h] = q[h];
}

// ---- Kernel 2: fused window attention -------------------------------------
// Dynamic LDS layout (bytes):
//   tokOt [128*64] f16  0..16384      | Wq [192*64] f16 16384..40960
//   Qs [128*64] 40960..57344          | Ks [128*64] 57344..73728
//   Vt [64*128] 73728..90112          | S  [128*128] f32 90112..155648
//   P  [128*128] f16 155648..188416   | ptab 188416..195856
//   bqkv 195856..196624               | bproj 196624..196880
// TDM stage buffer (128 tok x 64 ch f32 = 32KB) aliases S (consumed pre-S).
#define ATTN_LDS 196880

__global__ __launch_bounds__(256) void attn_kernel(
    const float* __restrict__ x, const float* __restrict__ y,
    const float* __restrict__ n1g, const float* __restrict__ n1b,
    const float* __restrict__ qkv_w, const float* __restrict__ qkv_b,
    const float* __restrict__ proj_w, const float* __restrict__ proj_b,
    const float* __restrict__ ptab_g, float* __restrict__ out) {
  extern __shared__ char smem[];
  _Float16* tokOt = (_Float16*)smem;            // 128x64
  _Float16* Wq    = tokOt + 128 * 64;           // 192x64 (Bt layout)
  _Float16* Qs    = Wq + 192 * 64;              // 128x64
  _Float16* Ks    = Qs + 128 * 64;              // 128x64
  _Float16* Vt    = Ks + 128 * 64;              // 64x128 (V^T)
  float*    S     = (float*)(Vt + 64 * 128);    // 128x128
  _Float16* P     = (_Float16*)(S + 128 * 128); // 128x128
  float*    ptab  = (float*)(P + 128 * 128);    // 465*4
  float*    bqkv  = ptab + 465 * 4;             // 192
  float*    bproj = bqkv + 192;                 // 64
  float*    stage = S;                          // TDM stage alias (32KB)

  const int t = threadIdx.x, lane = t & 31, wave = t >> 5;
  const int wid = blockIdx.x;
  const int b = wid >> 10, widx = wid & 1023;
  const int wr = widx >> 5, wc = widx & 31;
  const int hi8 = (lane & 16) ? 8 : 0;

  const unsigned long long winByte =
      ((unsigned long long)b * 65536ull + (unsigned long long)(wr * 8) * 256ull +
       (unsigned long long)(wc * 8)) * 64ull * 4ull;

#if HAVE_TDM
  // Async DMA: 8 rows x 512 f32 (8 win-cols x 64 ch), row stride = W*C.
  if (wave == 0) {
    unsigned lbase = lds_offset_of(stage);
    tdm_load_2d(lbase,         (unsigned long long)(uintptr_t)x + winByte,
                512u, 8u, 16384ull);
    tdm_load_2d(lbase + 16384u, (unsigned long long)(uintptr_t)y + winByte,
                512u, 8u, 16384ull);
  }
#endif

  __builtin_prefetch(qkv_w + t, 0, 1);
  __builtin_prefetch(proj_w + t, 0, 1);

  // ---- stage weights / biases / bias-table into LDS (Bt layout = [n][k])
  for (int i = t; i < 192 * 64; i += 256) {
    int n = i >> 6, k = i & 63;
    Wq[n * 64 + k] = (_Float16)qkv_w[k * 192 + n];
  }
  for (int i = t; i < 192; i += 256) bqkv[i] = qkv_b[i];
  for (int i = t; i < 64; i += 256) bproj[i] = proj_b[i];
  for (int i = t; i < 465 * 4; i += 256) ptab[i] = ptab_g[i];

#if HAVE_TDM
  if (wave == 0) __builtin_amdgcn_s_wait_tensorcnt(0);
  __syncthreads();  // TDM stage + weights visible to all waves
#endif

  // ---- LN1 of the 128 window tokens -> tokOt (f16)
  if (t < 128) {
#if HAVE_TDM
    const float* row = stage + t * 64;  // token-major in LDS stage
#else
    int mo_ = t >> 6, li_ = t & 63, lr_ = li_ >> 3, lc_ = li_ & 7;
    const float* src_ = mo_ ? y : x;
    const float* row =
        src_ + ((size_t)b * 65536 + (size_t)(wr * 8 + lr_) * 256 +
                (wc * 8 + lc_)) * 64;
#endif
    float mu = 0.f;
    for (int c = 0; c < 64; ++c) mu += row[c];
    mu *= (1.f / 64.f);
    float var = 0.f;
    for (int c = 0; c < 64; ++c) { float d = row[c] - mu; var += d * d; }
    var *= (1.f / 64.f);
    float inv = rsqrtf(var + 1e-5f);
    for (int c = 0; c < 64; ++c)
      tokOt[t * 64 + c] = (_Float16)((row[c] - mu) * inv * n1g[c] + n1b[c]);
  }
  __syncthreads();

  // ---- QKV GEMM: (128x64)@(64x192). 96 tiles, 12 per wave, K-loop = 2.
  for (int tile = wave; tile < 96; tile += 8) {
    int mi = tile / 12, ni = tile % 12;
    v8f acc = {};
#pragma unroll
    for (int ks = 0; ks < 2; ++ks) {
      v16h a = frag_a(tokOt, 64, mi * 16, ks * 32, 64, lane);
      v16h bb = frag_b(Wq, 64, ni * 16, ks * 32, 64, lane);
      acc = wmma_f16(a, bb, acc);
    }
    int n = ni * 16 + (lane & 15);
    if (n < 128) {                            // Q or K
      float scale = (n < 64) ? 0.25f : 1.f;   // d^-0.5 = 1/4 on Q
      _Float16* dst = (n < 64) ? Qs : Ks;
      int nn = n & 63;
      float bias = bqkv[n];
#pragma unroll
      for (int j = 0; j < 8; ++j) {
        int m = mi * 16 + j + hi8;
        dst[m * 64 + nn] = (_Float16)((acc[j] + bias) * scale);
      }
    } else {                                  // V -> transposed store
      int ch = n - 128;
      float bias = bqkv[n];
      v8h pk;
#pragma unroll
      for (int j = 0; j < 8; ++j) pk[j] = (_Float16)(acc[j] + bias);
      *(v8h*)(Vt + ch * 128 + mi * 16 + hi8) = pk;  // 16B ds_store
    }
  }
  __syncthreads();

  // reuse Wq region for proj_w^T (64x64)
  for (int i = t; i < 64 * 64; i += 256) {
    int n = i >> 6, k = i & 63;
    Wq[n * 64 + k] = (_Float16)proj_w[k * 64 + n];
  }

  // ---- per-head attention (heads sequential, S/P buffers reused)
  for (int h = 0; h < 4; ++h) {
    // S = Q K^T + bias ; 64 tiles, 8 per wave, d=16 zero-padded to K=32
    for (int tile = wave; tile < 64; tile += 8) {
      int mi = tile >> 3, ni = tile & 7;
      v16h a = frag_a(Qs, 64, mi * 16, h * 16, 16, lane);
      v16h bb = frag_b(Ks, 64, ni * 16, h * 16, 16, lane);
      v8f acc = {};
      acc = wmma_f16(a, bb, acc);
      int n = ni * 16 + (lane & 15);
      int rn = n >> 3, cn = n & 7;
#pragma unroll
      for (int j = 0; j < 8; ++j) {
        int m = mi * 16 + j + hi8;
        int rm = m >> 3, cm = m & 7;
        int idx = (rm - rn + 15) * 15 + (cm - cn + 7);
        S[m * 128 + n] = acc[j] + ptab[idx * 4 + h];
      }
    }
    __syncthreads();

    // row softmax -> P (f16)
    if (t < 128) {
      float mx = -1e30f;
      for (int n = 0; n < 128; ++n) mx = fmaxf(mx, S[t * 128 + n]);
      float sum = 0.f;
      for (int n = 0; n < 128; ++n) {
        float e = __expf(S[t * 128 + n] - mx);
        sum += e;
        P[t * 128 + n] = (_Float16)e;
      }
      float inv = 1.f / sum;
      for (int n = 0; n < 128; ++n)
        P[t * 128 + n] = (_Float16)((float)P[t * 128 + n] * inv);
    }
    __syncthreads();

    // O_h = P @ V_h : 8 M-tiles (one per wave), K-loop = 4 over 128 tokens
    {
      int mi = wave;
      v8f acc = {};
#pragma unroll
      for (int ks = 0; ks < 4; ++ks) {
        v16h a = frag_a(P, 128, mi * 16, ks * 32, 128, lane);
        v16h bb = frag_b(Vt, 128, h * 16, ks * 32, 128, lane);
        acc = wmma_f16(a, bb, acc);
      }
      int nn = h * 16 + (lane & 15);
#pragma unroll
      for (int j = 0; j < 8; ++j) {
        int m = mi * 16 + j + hi8;
        tokOt[m * 64 + nn] = (_Float16)acc[j];
      }
    }
    __syncthreads();
  }

  // ---- proj + residual + scatter to d_out. 32 tiles, 4 per wave.
  for (int tile = wave; tile < 32; tile += 8) {
    int mi = tile >> 2, ni = tile & 3;
    v8f acc = {};
#pragma unroll
    for (int ks = 0; ks < 2; ++ks) {
      v16h a = frag_a(tokOt, 64, mi * 16, ks * 32, 64, lane);
      v16h bb = frag_b(Wq, 64, ni * 16, ks * 32, 64, lane);
      acc = wmma_f16(a, bb, acc);
    }
    int ch = ni * 16 + (lane & 15);
    float bias = bproj[ch];
#pragma unroll
    for (int j = 0; j < 8; ++j) {
      int m = mi * 16 + j + hi8;
      int mo = m >> 6, li = m & 63, lr = li >> 3, lc = li & 7;
      size_t flat =
          ((size_t)b * 65536 + (size_t)(wr * 8 + lr) * 256 + (wc * 8 + lc)) *
              64 + ch;
      const float* src = mo ? y : x;
      out[(size_t)mo * OUTHALF + flat] = acc[j] + bias + src[flat];
    }
  }
}

// ---- Kernel 3: fused FFN (LN2 -> 64->256 GELU -> 64, + residual) ----------
// One block per 128 tokens; blocks [0,1024) = x, [1024,2048) = y.
// Dynamic LDS: xn 16384 | W1t 32768 | h 65536 | W2t 32768 | b1 1024 | b2 256
// TDM stage (32KB contiguous token slab) aliases h (consumed pre-GEMM1).
#define FFN_LDS 148736

__global__ __launch_bounds__(256) void ffn_kernel(
    float* __restrict__ out, const float* __restrict__ n2g,
    const float* __restrict__ n2b, const float* __restrict__ fx1_w,
    const float* __restrict__ fx1_b, const float* __restrict__ fx2_w,
    const float* __restrict__ fx2_b, const float* __restrict__ fy1_w,
    const float* __restrict__ fy1_b, const float* __restrict__ fy2_w,
    const float* __restrict__ fy2_b) {
  extern __shared__ char smem[];
  _Float16* xn  = (_Float16*)smem;        // 128x64
  _Float16* W1t = xn + 128 * 64;          // 256x64  (Bt layout)
  _Float16* hb  = W1t + 256 * 64;         // 128x256 hidden
  _Float16* W2t = hb + 128 * 256;         // 64x256  (Bt layout)
  float*    b1  = (float*)(W2t + 64 * 256);
  float*    b2  = b1 + 256;
  float*    stage = (float*)hb;           // TDM stage alias (32KB)

  const int t = threadIdx.x, lane = t & 31, wave = t >> 5;
  const int hi8 = (lane & 16) ? 8 : 0;
  const int bb_ = blockIdx.x;
  const int mo = bb_ >> 10;
  const int tok0 = (bb_ & 1023) * 128;
  const float* w1 = mo ? fy1_w : fx1_w;
  const float* wb1 = mo ? fy1_b : fx1_b;
  const float* w2 = mo ? fy2_w : fx2_w;
  const float* wb2 = mo ? fy2_b : fx2_b;
  float* base = out + (size_t)mo * OUTHALF;

#if HAVE_TDM
  // Async DMA: contiguous 8192-f32 slab (128 tokens x 64 ch) -> LDS stage.
  if (wave == 0) {
    tdm_load_2d(lds_offset_of(stage),
                (unsigned long long)(uintptr_t)base +
                    (unsigned long long)tok0 * 64ull * 4ull,
                8192u, 1u, 8192ull);
  }
#endif

  __builtin_prefetch(w1 + t, 0, 1);
  __builtin_prefetch(w2 + t, 0, 1);

  // stage weights (transposed) + biases
  for (int i = t; i < 256 * 64; i += 256) {   // W1t[n][k] = w1[k*256+n]
    int n = i >> 6, k = i & 63;
    W1t[n * 64 + k] = (_Float16)w1[k * 256 + n];
  }
  for (int i = t; i < 64 * 256; i += 256) {   // W2t[n][k] = w2[k*64+n]
    int n = i >> 8, k = i & 255;
    W2t[n * 256 + k] = (_Float16)w2[k * 64 + n];
  }
  for (int i = t; i < 256; i += 256) b1[i] = wb1[i];
  for (int i = t; i < 64; i += 256) b2[i] = wb2[i];

#if HAVE_TDM
  if (wave == 0) __builtin_amdgcn_s_wait_tensorcnt(0);
  __syncthreads();
#endif

  // LN2 of 128 tokens (input = current out buffer, also the residual)
  if (t < 128) {
#if HAVE_TDM
    const float* row = stage + t * 64;
#else
    const float* row = base + (size_t)(tok0 + t) * 64;
#endif
    float mu = 0.f;
    for (int c = 0; c < 64; ++c) mu += row[c];
    mu *= (1.f / 64.f);
    float var = 0.f;
    for (int c = 0; c < 64; ++c) { float d = row[c] - mu; var += d * d; }
    var *= (1.f / 64.f);
    float inv = rsqrtf(var + 1e-5f);
    for (int c = 0; c < 64; ++c)
      xn[t * 64 + c] = (_Float16)((row[c] - mu) * inv * n2g[c] + n2b[c]);
  }
  __syncthreads();

  // GEMM1: (128x64)@(64x256) -> GELU -> h. 128 tiles, 16/wave, K-loop = 2.
  for (int tile = wave; tile < 128; tile += 8) {
    int mi = tile >> 4, ni = tile & 15;
    v8f acc = {};
#pragma unroll
    for (int ks = 0; ks < 2; ++ks) {
      v16h a = frag_a(xn, 64, mi * 16, ks * 32, 64, lane);
      v16h bbv = frag_b(W1t, 64, ni * 16, ks * 32, 64, lane);
      acc = wmma_f16(a, bbv, acc);
    }
    int n = ni * 16 + (lane & 15);
    float bias = b1[n];
#pragma unroll
    for (int j = 0; j < 8; ++j) {
      int m = mi * 16 + j + hi8;
      float u = acc[j] + bias;
      float g = 0.5f * u * (1.f + erff(u * 0.70710678118654752f)); // exact GELU
      hb[m * 256 + n] = (_Float16)g;
    }
  }
  __syncthreads();

  // GEMM2: (128x256)@(256x64) + b2 + residual -> out. 32 tiles, K-loop = 8.
  for (int tile = wave; tile < 32; tile += 8) {
    int mi = tile >> 2, ni = tile & 3;
    v8f acc = {};
#pragma unroll
    for (int ks = 0; ks < 8; ++ks) {
      v16h a = frag_a(hb, 256, mi * 16, ks * 32, 256, lane);
      v16h bbv = frag_b(W2t, 256, ni * 16, ks * 32, 256, lane);
      acc = wmma_f16(a, bbv, acc);
    }
    int ch = ni * 16 + (lane & 15);
    float bias = b2[ch];
#pragma unroll
    for (int j = 0; j < 8; ++j) {
      int m = mi * 16 + j + hi8;
      size_t off = (size_t)(tok0 + m) * 64 + ch;
      base[off] = base[off] + acc[j] + bias;   // residual add, in-place
    }
  }
}

// ---------------------------------------------------------------------------

extern "C" void kernel_launch(void* const* d_in, const int* in_sizes, int n_in,
                              void* d_out, int out_size, void* d_ws,
                              size_t ws_size, hipStream_t stream) {
  (void)in_sizes; (void)n_in; (void)out_size; (void)ws_size;
  const float* x      = (const float*)d_in[0];
  const float* y      = (const float*)d_in[1];
  const float* n1g    = (const float*)d_in[2];
  const float* n1b    = (const float*)d_in[3];
  const float* qkv_w  = (const float*)d_in[4];
  const float* qkv_b  = (const float*)d_in[5];
  const float* proj_w = (const float*)d_in[6];
  const float* proj_b = (const float*)d_in[7];
  const float* pp_w   = (const float*)d_in[8];
  const float* pp_b   = (const float*)d_in[9];
  const float* ln1g   = (const float*)d_in[10];
  const float* ln1b   = (const float*)d_in[11];
  const float* l1w    = (const float*)d_in[12];
  const float* l1b    = (const float*)d_in[13];
  const float* ln2g   = (const float*)d_in[14];
  const float* ln2b   = (const float*)d_in[15];
  const float* l2w    = (const float*)d_in[16];
  const float* l2b    = (const float*)d_in[17];
  const float* ln3g   = (const float*)d_in[18];
  const float* ln3b   = (const float*)d_in[19];
  const float* l3w    = (const float*)d_in[20];
  const float* l3b    = (const float*)d_in[21];
  const float* n2g    = (const float*)d_in[22];
  const float* n2b    = (const float*)d_in[23];
  const float* fx1_w  = (const float*)d_in[24];
  const float* fx1_b  = (const float*)d_in[25];
  const float* fx2_w  = (const float*)d_in[26];
  const float* fx2_b  = (const float*)d_in[27];
  const float* fy1_w  = (const float*)d_in[28];
  const float* fy1_b  = (const float*)d_in[29];
  const float* fy2_w  = (const float*)d_in[30];
  const float* fy2_b  = (const float*)d_in[31];
  float* out  = (float*)d_out;
  float* ptab = (float*)d_ws;  // 465*4 floats

  posbias_kernel<<<1, 512, 0, stream>>>(pp_w, pp_b, ln1g, ln1b, l1w, l1b,
                                        ln2g, ln2b, l2w, l2b, ln3g, ln3b,
                                        l3w, l3b, ptab);
  attn_kernel<<<NWIN, 256, ATTN_LDS, stream>>>(x, y, n1g, n1b, qkv_w, qkv_b,
                                               proj_w, proj_b, ptab, out);
  ffn_kernel<<<NWIN, 256, FFN_LDS, stream>>>(out, n2g, n2b, fx1_w, fx1_b,
                                             fx2_w, fx2_b, fy1_w, fy1_b,
                                             fy2_w, fy2_b);
}
